// Sinkhorn_Decoder_34832184770747
// MI455X (gfx1250) — compile-verified, matrix operand order checked by reading, not run on hardware
//
#include <hip/hip_runtime.h>
#include <hip/hip_fp16.h>

typedef __attribute__((ext_vector_type(16))) _Float16 v16h;
typedef __attribute__((ext_vector_type(8)))  float    v8f;

#define B_        64
#define N_        128
#define BN_       8192
#define EWO_PER_B (N_ * (N_ - 1))      /* 16256 */
#define EWO_TOT   (B_ * EWO_PER_B)     /* 1040384 */

// ---------------------------------------------------------------------------
// Head: number_nodes MLP (64->64->32->1, leaky-relu) + latent_transform emb.
// Intermediates in LDS, rolled loops -> compact code, no scratch.
// ---------------------------------------------------------------------------
__global__ void head_kernel(const float* __restrict__ latent,
                            const float* __restrict__ ltW1, const float* __restrict__ ltW2,
                            const float* __restrict__ nnW1, const float* __restrict__ nnb1,
                            const float* __restrict__ nnW2, const float* __restrict__ nnb2,
                            const float* __restrict__ nnW3, const float* __restrict__ nnb3,
                            float* __restrict__ emb, float* __restrict__ nn_out)
{
  __shared__ float h1s[64 * 64];
  __shared__ float h2s[64 * 32];
  __shared__ float e1s[64 * 32];
  int b = threadIdx.x;
  if (b >= B_) return;
  const float* lat = latent + b * 64;
#pragma unroll 1
  for (int o = 0; o < 64; ++o) {
    float s = nnb1[o];
#pragma unroll 1
    for (int i = 0; i < 64; ++i) s += lat[i] * nnW1[i * 64 + o];
    h1s[b * 64 + o] = s >= 0.f ? s : 0.01f * s;
  }
#pragma unroll 1
  for (int o = 0; o < 32; ++o) {
    float s = nnb2[o];
#pragma unroll 1
    for (int i = 0; i < 64; ++i) s += h1s[b * 64 + i] * nnW2[i * 32 + o];
    h2s[b * 32 + o] = s >= 0.f ? s : 0.01f * s;
  }
  float nn = nnb3[0];
#pragma unroll 1
  for (int i = 0; i < 32; ++i) nn += h2s[b * 32 + i] * nnW3[i];
  nn_out[b] = nn;
#pragma unroll 1
  for (int o = 0; o < 32; ++o) {
    float s = 0.f;
#pragma unroll 1
    for (int i = 0; i < 64; ++i) s += lat[i] * ltW1[i * 32 + o];
    e1s[b * 32 + o] = s >= 0.f ? s : 0.01f * s;
  }
#pragma unroll 1
  for (int o = 0; o < 16; ++o) {
    float s = 0.f;
#pragma unroll 1
    for (int i = 0; i < 32; ++i) s += e1s[b * 32 + i] * ltW2[i * 16 + o];
    emb[b * 16 + o] = s;
  }
}

// ---------------------------------------------------------------------------
// Fuse mW[d,out] and eW[d,5] into padded f16 Wc[32 cols][64 k] per layer,
// and fused bias f32[32] per layer.
// ---------------------------------------------------------------------------
struct LayerParams { const float* mW; const float* mb; const float* eW; const float* eb; };
struct AllLayerParams { LayerParams L[8]; };

__global__ void prep_weights(AllLayerParams P, _Float16* __restrict__ wbuf,
                             float* __restrict__ bbuf)
{
  int L   = blockIdx.x;                 // 0..7
  int d   = (L & 1) ? 37 : 45;
  int out = (L & 1) ? 4  : 16;
  const LayerParams lp = P.L[L];
#pragma unroll 1
  for (int idx = threadIdx.x; idx < 32 * 64; idx += blockDim.x) {
    int n = idx >> 6, k = idx & 63;
    float v = 0.f;
    if (k < d) {
      if (n < out)          v = lp.mW[k * out + n];
      else if (n < out + 5) v = lp.eW[k * 5 + (n - out)];
    }
    wbuf[L * 2048 + idx] = (_Float16)v;
  }
  for (int n = threadIdx.x; n < 32; n += blockDim.x) {
    float v = 0.f;
    if (n < out)          v = lp.mb[n];
    else if (n < out + 5) v = lp.eb[n - out];
    bbuf[L * 32 + n] = v;
  }
}

// ---------------------------------------------------------------------------
// x0 = [emb(16) | noise(4)]  and self-loop attr = [0,1,0,0,0]
// ---------------------------------------------------------------------------
__global__ void init_x_sl(const float* __restrict__ emb, const float* __restrict__ noise,
                          float* __restrict__ x20, float* __restrict__ sl0)
{
  int node = blockIdx.x * blockDim.x + threadIdx.x;
  if (node >= BN_) return;
  int b = node >> 7;
#pragma unroll
  for (int c = 0; c < 16; ++c) x20[node * 20 + c]      = emb[b * 16 + c];
#pragma unroll
  for (int c = 0; c < 4;  ++c) x20[node * 20 + 16 + c] = noise[node * 4 + c];
#pragma unroll
  for (int c = 0; c < 5;  ++c) sl0[node * 5 + c]       = (c == 1) ? 1.f : 0.f;
}

__global__ void zero_buf4(float4* __restrict__ p, int n4)
{
  int i = blockIdx.x * blockDim.x + threadIdx.x;
  if (i < n4) p[i] = make_float4(0.f, 0.f, 0.f, 0.f);
}

// ---------------------------------------------------------------------------
// WMMA fragment load: 16 halves per lane per 16x32 chunk, two ds_load_b128.
// ---------------------------------------------------------------------------
__device__ inline v16h ld_frag16(const _Float16* base)
{
  union { float4 f; _Float16 h[8]; } u0, u1;
  u0.f = *(const float4*)(base);
  u1.f = *(const float4*)(base + 16);
  v16h r;
#pragma unroll
  for (int i = 0; i < 8; ++i) { r[i] = u0.h[i]; r[i + 8] = u1.h[i]; }
  return r;
}

// ---------------------------------------------------------------------------
// One wave32 per (batch,dst): 128 in-edges -> 8 tiles of 16 edges.
// Per tile: D = A(16 edges x 64k, f16) x Wc(64k x 32n, f16) via
// v_wmma_f32_16x16x32_f16; bias+relu on the f32 accumulator; message columns
// reduced in-wave (shfl_xor over wave32, zero global atomics); edge-attr
// columns scattered through a per-edge pointer table (no per-row branching).
// Staging is branch-once-on-lane-parity with straight-line, compile-time
// indexed segments committed as 4x ds_store_b128.
// ---------------------------------------------------------------------------
template<int DIN, int OUT, int NC>
__launch_bounds__(32)
__global__ void gnn_layer(const float* __restrict__ x_in,
                          float* __restrict__ x_out,
                          const float* __restrict__ attr_in,
                          float* __restrict__ attr_out,
                          const float* __restrict__ sl_in,
                          float* __restrict__ sl_out,
                          const _Float16* __restrict__ wc,   // [32][64] f16
                          const float* __restrict__ bias)    // [32]
{
  constexpr int D = 2 * DIN + 5;
  __shared__ __align__(16) _Float16 zs[16 * 64];
  __shared__ __align__(16) _Float16 ws[32 * 64];
  __shared__ __align__(16) float xj[DIN];
  __shared__ float* aoptr[16];         // per-edge attr output pointer

  const int wg = blockIdx.x;
  const int b = wg >> 7, j = wg & 127, node = wg;
  const int lane = threadIdx.x;
  const int hi = lane >> 4, lm = lane & 15;

  if (lane < DIN) xj[lane] = x_in[(size_t)node * DIN + lane];
  {
    const float4* wsrc = (const float4*)wc;
    float4* wdst = (float4*)ws;
#pragma unroll
    for (int i = 0; i < 8; ++i) wdst[lane + i * 32] = wsrc[lane + i * 32];
  }
  __syncthreads();

  // B fragments: lane holds column n = nc*16+lm, K pattern mirrors A layout.
  v16h bf[2][NC];
#pragma unroll
  for (int kc = 0; kc < 2; ++kc)
#pragma unroll
    for (int nc = 0; nc < NC; ++nc)
      bf[kc][nc] = ld_frag16(&ws[(nc * 16 + lm) * 64 + kc * 32 + hi * 8]);

  float bcol[NC];
#pragma unroll
  for (int nc = 0; nc < NC; ++nc) bcol[nc] = bias[nc * 16 + lm];

  float xacc[NC];
#pragma unroll
  for (int nc = 0; nc < NC; ++nc) xacc[nc] = 0.f;

  const int ewoBase = b * EWO_PER_B;
  const int et = lane >> 1;            // edge row this lane helps stage
  const int p  = lane & 1;             // which 32-wide K half it fills

  for (int t = 0; t < 8; ++t) {
    int slot = t * 16 + et;            // 0..127 in-edge slot for dst j
    int srcN, eid, isSL;
    if (slot < 127) {
      int i = slot + (slot >= j ? 1 : 0);          // src node (skip i==j)
      srcN = b * 128 + i;
      eid  = ewoBase + i * 127 + j - (j > i ? 1 : 0);
      isSL = 0;
    } else {
      srcN = node; eid = 0; isSL = 1;              // appended self-loop
    }
    __syncthreads();                    // prior tile fully consumed

    union { float4 q[4]; _Float16 h[32]; } seg;
    if (p == 0) {
      aoptr[et] = isSL ? (sl_out + (size_t)node * 5)
                       : (attr_out + (size_t)eid * 5);
      // K = 0..31 : x_src[0..DIN) then x_dst[0..32-DIN)
      const float4* xr4 = (const float4*)(x_in + (size_t)srcN * DIN);
#pragma unroll
      for (int q = 0; q < DIN / 4; ++q) {
        float4 f = xr4[q];
        seg.h[4 * q + 0] = (_Float16)f.x;
        seg.h[4 * q + 1] = (_Float16)f.y;
        seg.h[4 * q + 2] = (_Float16)f.z;
        seg.h[4 * q + 3] = (_Float16)f.w;
      }
#pragma unroll
      for (int k = DIN; k < 32; ++k) seg.h[k] = (_Float16)xj[k - DIN];
    } else {
      // K = 32..63 : rest of x_dst, 5 edge attrs, zero pad
      const float* ap = isSL ? (sl_in + (size_t)node * 5)
                             : (attr_in + (size_t)eid * 5);
#pragma unroll
      for (int k = 32; k < 2 * DIN; ++k) seg.h[k - 32] = (_Float16)xj[k - DIN];
#pragma unroll
      for (int k = 0; k < 5; ++k) seg.h[2 * DIN - 32 + k] = (_Float16)ap[k];
#pragma unroll
      for (int k = D - 32; k < 32; ++k) seg.h[k] = (_Float16)0.f;
    }
    {
      float4* dst = (float4*)&zs[et * 64 + p * 32];
#pragma unroll
      for (int i = 0; i < 4; ++i) dst[i] = seg.q[i];
    }
    __syncthreads();

    v8f c[NC];
#pragma unroll
    for (int nc = 0; nc < NC; ++nc)
#pragma unroll
      for (int r = 0; r < 8; ++r) c[nc][r] = 0.f;

#pragma unroll
    for (int kc = 0; kc < 2; ++kc) {
      v16h a = ld_frag16(&zs[lm * 64 + kc * 32 + hi * 8]);
#pragma unroll
      for (int nc = 0; nc < NC; ++nc)
        c[nc] = __builtin_amdgcn_wmma_f32_16x16x32_f16(
                    false, a, false, bf[kc][nc], (short)0, c[nc], false, false);
    }

    // bias + relu; accumulate messages per column; scatter edge attrs.
#pragma unroll
    for (int nc = 0; nc < NC; ++nc) {
      int n = nc * 16 + lm;
      if (n < OUT) {
        float s = 0.f;
#pragma unroll
        for (int r = 0; r < 8; ++r) {
          float v = c[nc][r] + bcol[nc];
          s += (v > 0.f ? v : 0.f);
        }
        xacc[nc] += s;
      } else if (n < OUT + 5) {
#pragma unroll
        for (int r = 0; r < 8; ++r) {
          float v = c[nc][r] + bcol[nc];
          v = v > 0.f ? v : 0.f;
          aoptr[hi * 8 + r][n - OUT] = v;
        }
      }
    }
  }

  // combine row halves (lanes L and L+16 hold same column) and store x_new
#pragma unroll
  for (int nc = 0; nc < NC; ++nc) {
    float tot = xacc[nc] + __shfl_xor(xacc[nc], 16, 32);
    int n = nc * 16 + lm;
    if (hi == 0 && n < OUT)
      x_out[(size_t)node * OUT + n] = tot;
  }
}

// ---------------------------------------------------------------------------
// PairNorm + ReLU, dim 16 (in place)
// ---------------------------------------------------------------------------
__global__ void pairnorm_relu16(float* __restrict__ x)
{
  __shared__ float cs[16];
  __shared__ float var;
  int t = threadIdx.x;
  int node = blockIdx.x * 128 + t;
  float v[16];
#pragma unroll
  for (int c = 0; c < 16; ++c) v[c] = x[(size_t)node * 16 + c];
  if (t < 16) cs[t] = 0.f;
  if (t == 0) var = 0.f;
  __syncthreads();
#pragma unroll
  for (int c = 0; c < 16; ++c) atomicAdd(&cs[c], v[c]);
  __syncthreads();
  float sq = 0.f, xc[16];
#pragma unroll
  for (int c = 0; c < 16; ++c) { xc[c] = v[c] - cs[c] * (1.f / 128.f); sq += xc[c] * xc[c]; }
  atomicAdd(&var, sq);
  __syncthreads();
  float scale = rsqrtf(var * (1.f / 128.f) + 1e-5f);
#pragma unroll
  for (int c = 0; c < 16; ++c) {
    float u = xc[c] * scale;
    x[(size_t)node * 16 + c] = u > 0.f ? u : 0.f;
  }
}

// ---------------------------------------------------------------------------
// PairNorm + ReLU dim 4; writes stage output and next-stage [x | emb] concat
// ---------------------------------------------------------------------------
__global__ void pairnorm_relu4_concat(const float* __restrict__ xin,
                                      const float* __restrict__ emb,
                                      float* __restrict__ outs,
                                      float* __restrict__ x20)
{
  __shared__ float cs[4];
  __shared__ float var;
  int t = threadIdx.x;
  int bidx = blockIdx.x;
  int node = bidx * 128 + t;
  float v[4];
#pragma unroll
  for (int c = 0; c < 4; ++c) v[c] = xin[(size_t)node * 4 + c];
  if (t < 4) cs[t] = 0.f;
  if (t == 0) var = 0.f;
  __syncthreads();
#pragma unroll
  for (int c = 0; c < 4; ++c) atomicAdd(&cs[c], v[c]);
  __syncthreads();
  float sq = 0.f, xc[4];
#pragma unroll
  for (int c = 0; c < 4; ++c) { xc[c] = v[c] - cs[c] * (1.f / 128.f); sq += xc[c] * xc[c]; }
  atomicAdd(&var, sq);
  __syncthreads();
  float scale = rsqrtf(var * (1.f / 128.f) + 1e-5f);
#pragma unroll
  for (int c = 0; c < 4; ++c) {
    float u = xc[c] * scale; u = u > 0.f ? u : 0.f;
    outs[(size_t)node * 4 + c] = u;
    x20[(size_t)node * 20 + c] = u;
  }
#pragma unroll
  for (int c = 0; c < 16; ++c) x20[(size_t)node * 20 + 4 + c] = emb[bidx * 16 + c];
}

// ---------------------------------------------------------------------------
extern "C" void kernel_launch(void* const* d_in, const int* in_sizes, int n_in,
                              void* d_out, int out_size, void* d_ws, size_t ws_size,
                              hipStream_t stream)
{
  (void)out_size; (void)ws_size;

  // param leaf sizes, insertion order
  static const int psz[40] = {2048, 512, 4096, 64, 2048, 32, 32, 1,
                              720, 16, 225, 5, 148, 4, 185, 5,
                              720, 16, 225, 5, 148, 4, 185, 5,
                              720, 16, 225, 5, 148, 4, 185, 5,
                              720, 16, 225, 5, 148, 4, 185, 5};
  // P[0]=lt_W1 P[1]=lt_W2 P[2]=nn_W1 P[3]=nn_b1 P[4]=nn_W2 P[5]=nn_b2
  // P[6]=nn_W3 P[7]=nn_b3; P[8+4L+{0,1,2,3}] = g{s}{l}_{mW,mb,eW,eb}
  const float* latent = (const float*)d_in[0];
  const float* noise  = (const float*)d_in[1];
  const float* P[40];

  if (n_in >= 45) {
    if (in_sizes[2] == 2048) {               // insertion-order flattened leaves
      for (int i = 0; i < 40; ++i) P[i] = (const float*)d_in[2 + i];
    } else {                                 // jax sorted-key tree order
      int base = 2;
      for (int L = 0; L < 8; ++L) {
        P[8 + 4 * L + 2] = (const float*)d_in[base++]; // eW
        P[8 + 4 * L + 3] = (const float*)d_in[base++]; // eb
        P[8 + 4 * L + 0] = (const float*)d_in[base++]; // mW
        P[8 + 4 * L + 1] = (const float*)d_in[base++]; // mb
      }
      P[0] = (const float*)d_in[base++]; P[1] = (const float*)d_in[base++];
      P[2] = (const float*)d_in[base++]; P[4] = (const float*)d_in[base++];
      P[6] = (const float*)d_in[base++]; P[3] = (const float*)d_in[base++];
      P[5] = (const float*)d_in[base++]; P[7] = (const float*)d_in[base++];
    }
  } else {                                   // params passed as one blob
    const float* blob = (const float*)d_in[2];
    size_t off = 0;
    for (int i = 0; i < 40; ++i) { P[i] = blob + off; off += (size_t)psz[i]; }
  }

  // workspace layout (floats)
  float* wsf     = (float*)d_ws;
  float* emb     = wsf;                       // 1024
  float* x20     = emb + 1024;                // 163840
  float* xtmp    = x20 + 163840;              // 131072
  float* xsmall  = xtmp + 131072;             // 32768
  float* sl0     = xsmall + 32768;            // 40960
  float* sl1     = sl0 + 40960;               // 40960
  float* bbuf    = sl1 + 40960;               // 256
  _Float16* wbuf = (_Float16*)(bbuf + 256);   // 8*2048 halves

  // output layout
  float* out        = (float*)d_out;
  float* outs_base  = out;                                    // [4][8192][4]
  float* aouts_base = out + 4 * (size_t)BN_ * 4;              // [4][EWO_TOT][5]
  float* R[4];
  for (int s = 0; s < 4; ++s) R[s] = aouts_base + (size_t)s * EWO_TOT * 5;
  float* nn_out = aouts_base + (size_t)4 * EWO_TOT * 5;       // [64]

  AllLayerParams alp;
  for (int L = 0; L < 8; ++L) {
    alp.L[L].mW = P[8 + 4 * L + 0]; alp.L[L].mb = P[8 + 4 * L + 1];
    alp.L[L].eW = P[8 + 4 * L + 2]; alp.L[L].eb = P[8 + 4 * L + 3];
  }

  prep_weights<<<dim3(8), dim3(64), 0, stream>>>(alp, wbuf, bbuf);
  head_kernel<<<dim3(1), dim3(64), 0, stream>>>(latent, P[0], P[1], P[2], P[3],
                                                P[4], P[5], P[6], P[7], emb, nn_out);
  init_x_sl<<<dim3(BN_ / 256), dim3(256), 0, stream>>>(emb, noise, x20, sl0);
  {
    int n4 = (EWO_TOT * 5) / 4;
    zero_buf4<<<dim3((n4 + 255) / 256), dim3(256), 0, stream>>>((float4*)R[0], n4);
  }

  float* sl[2] = { sl0, sl1 };
  for (int s = 0; s < 4; ++s) {
    int L0 = 2 * s, L1 = 2 * s + 1;
    const float* ain0 = (s == 0) ? R[0] : R[s - 1];
    gnn_layer<20, 16, 2><<<dim3(BN_), dim3(32), 0, stream>>>(
        x20, xtmp, ain0, R[s], sl[L0 & 1], sl[(L0 + 1) & 1],
        wbuf + L0 * 2048, bbuf + L0 * 32);
    pairnorm_relu16<<<dim3(B_), dim3(128), 0, stream>>>(xtmp);

    float* xo = (s == 3) ? (outs_base + (size_t)3 * BN_ * 4) : xsmall;
    gnn_layer<16, 4, 1><<<dim3(BN_), dim3(32), 0, stream>>>(
        xtmp, xo, R[s], R[s], sl[L1 & 1], sl[(L1 + 1) & 1],
        wbuf + L1 * 2048, bbuf + L1 * 32);
    if (s < 3)
      pairnorm_relu4_concat<<<dim3(B_), dim3(128), 0, stream>>>(
          xsmall, emb, outs_base + (size_t)s * BN_ * 4, x20);
  }
}
